// BoxMultiHeadedAttention_18932215840830
// MI455X (gfx1250) — matrix operand order repaired and checked by reading, not compiled, not run on hardware
//
#include <hip/hip_runtime.h>

// ---------------------------------------------------------------------------
// CDNA5 / gfx1250 BoxMultiHeadedAttention
//   B=16, N=256, D_MODEL=512, H=8, D_K=64
//   All matrix math through v_wmma_f32_16x16x32_bf16 (fp32 accumulate).
//   Weights pre-transposed to [N,K] bf16 so both A and B WMMA fragments are
//   contiguous 32B-per-lane-half loads (global_load_b128).
// ---------------------------------------------------------------------------

typedef __attribute__((ext_vector_type(16))) __bf16 v16bf;
typedef __attribute__((ext_vector_type(8)))  float  v8f;

union Frag {
  unsigned int u[8];
  v16bf        v;
};

__device__ __forceinline__ unsigned short f2bf(float f) {
  unsigned int u = __float_as_uint(f);
  u += 0x7FFFu + ((u >> 16) & 1u);   // round-to-nearest-even
  return (unsigned short)(u >> 16);
}

__device__ __forceinline__ v8f wmma_bf16(const Frag& a, const Frag& b, v8f c) {
  // (neg_a, A, neg_b, B, c_mod, C, reuse_a, reuse_b)
  return __builtin_amdgcn_wmma_f32_16x16x32_bf16(false, a.v, false, b.v,
                                                 (short)0, c, false, false);
}

// ---------------------------------------------------------------------------
// fp32 -> bf16 conversion (activations)
// ---------------------------------------------------------------------------
__global__ void f32_to_bf16_kernel(const float* __restrict__ in,
                                   unsigned short* __restrict__ out, int n) {
  int i = blockIdx.x * blockDim.x + threadIdx.x;
  if (i < n) out[i] = f2bf(in[i]);
}

// ---------------------------------------------------------------------------
// fp32 [512,512] -> bf16 transposed [512,512] (weights, col-major for B frags)
// ---------------------------------------------------------------------------
__global__ void f32_to_bf16_transpose_kernel(const float* __restrict__ in,
                                             unsigned short* __restrict__ out) {
  int i = blockIdx.x * blockDim.x + threadIdx.x;  // i = k*512 + c
  int k = i >> 9, c = i & 511;
  out[(size_t)c * 512 + k] = f2bf(in[i]);
}

// ---------------------------------------------------------------------------
// GEMM: C[4096,512] = A[4096,512](bf16) @ W + bias, W given as Wt[N,K] bf16.
//   mode 0: Q  -> [B*H,256,64] bf16, scale = 1/sqrt(64)
//   mode 1: K  -> [B*H,256,64] bf16
//   mode 2: V  -> [B*H,64,256] bf16 (transposed for attention B-fragments)
//   mode 3: O  -> [4096,512]   f32  (final output)
// One wave per 16x16 output tile; 16 WMMAs over K=512.
// Per K-chunk: A = 2x b128, B = 2x b128 (all contiguous per lane-half).
// ---------------------------------------------------------------------------
__global__ __launch_bounds__(32)
void gemm_wmma_kernel(const unsigned short* __restrict__ A,
                      const unsigned short* __restrict__ Wt,  // [512,512] col-major
                      const float* __restrict__ bias,
                      void* __restrict__ out, int mode, float scale) {
  const int lane = threadIdx.x & 31;
  const int half = lane >> 4;
  const int l16  = lane & 15;
  const int tn = blockIdx.x;  // 0..31 (cols / 16)
  const int tm = blockIdx.y;  // 0..255 (rows / 16)
  const int row = tm * 16 + l16;  // A-fragment row for this lane
  const int col = tn * 16 + l16;  // B-fragment column for this lane

  const unsigned short* arow = A  + (size_t)row * 512;
  const unsigned short* wcol = Wt + (size_t)col * 512;

  v8f acc{};
  for (int kc = 0; kc < 512; kc += 32) {
    Frag af, bf;
    // A 16x32 bf16 fragment: v0..3 -> K = 8*half + 2v, v4..7 -> 16 + 8*half + 2v
#pragma unroll
    for (int v = 0; v < 4; ++v)
      af.u[v] = *(const unsigned int*)(arow + kc + 8 * half + 2 * v);
#pragma unroll
    for (int v = 0; v < 4; ++v)
      af.u[4 + v] = *(const unsigned int*)(arow + kc + 16 + 8 * half + 2 * v);
    // B 32x16 bf16 fragment: v0..3 -> K = 8*half + 2v, v4..7 -> 16 + 8*half + 2v
    // Wt is col-major, so the pairs are contiguous.
#pragma unroll
    for (int v = 0; v < 4; ++v)
      bf.u[v] = *(const unsigned int*)(wcol + kc + 8 * half + 2 * v);
#pragma unroll
    for (int v = 0; v < 4; ++v)
      bf.u[4 + v] = *(const unsigned int*)(wcol + kc + 16 + 8 * half + 2 * v);
    if (kc + 64 < 512) {
      __builtin_prefetch(arow + kc + 64, 0, 3);  // global_prefetch_b8 (near)
      __builtin_prefetch(wcol + kc + 64, 0, 3);
    }
    acc = wmma_bf16(af, bf, acc);
  }

  // C/D layout: VGPR v -> row m = v + 8*half, col = lane%16
#pragma unroll
  for (int v = 0; v < 8; ++v) {
    int m = v + 8 * half;
    int r = tm * 16 + m;
    int c = tn * 16 + l16;
    float val = (acc[v] + bias[c]) * scale;
    if (mode == 3) {
      ((float*)out)[(size_t)r * 512 + c] = val;
    } else {
      int b = r >> 8, n = r & 255;
      int h = c >> 6, d = c & 63;
      unsigned short bv = f2bf(val);
      if (mode == 2)
        ((unsigned short*)out)[((size_t)(b * 8 + h) * 64 + d) * 256 + n] = bv;
      else
        ((unsigned short*)out)[((size_t)(b * 8 + h) * 256 + n) * 64 + d] = bv;
    }
  }
}

// ---------------------------------------------------------------------------
// Fused geometry bias: log(max(relu(w_g), 1e-6)) -> [B*H, 256, 256] f32
// One thread per (b, q, m) pair; 8 heads accumulated in registers.
// ---------------------------------------------------------------------------
__global__ __launch_bounds__(256)
void geo_bias_kernel(const float* __restrict__ box,
                     const float* __restrict__ WGw,
                     const float* __restrict__ WGb,
                     float* __restrict__ biasOut) {
  __shared__ float sW[8 * 64];
  __shared__ float sB[8];
  int tid = threadIdx.x;
  for (int i = tid; i < 512; i += 256) sW[i] = WGw[i];
  if (tid < 8) sB[tid] = WGb[tid];
  __syncthreads();

  int idx = blockIdx.x * 256 + tid;
  int b = idx >> 16;
  int q = (idx >> 8) & 255;
  int m = idx & 255;

  const float* bq_ = box + ((size_t)b * 256 + q) * 4;
  const float* bm_ = box + ((size_t)b * 256 + m) * 4;
  float cxq = (bq_[0] + bq_[2]) * 0.5f, cyq = (bq_[1] + bq_[3]) * 0.5f;
  float wq  = bq_[2] - bq_[0] + 1.0f,  hq  = bq_[3] - bq_[1] + 1.0f;
  float cxm = (bm_[0] + bm_[2]) * 0.5f, cym = (bm_[1] + bm_[3]) * 0.5f;
  float wm  = bm_[2] - bm_[0] + 1.0f,  hm  = bm_[3] - bm_[1] + 1.0f;

  float dl[4];
  dl[0] = __logf(fmaxf(fabsf((cxq - cxm) / wq), 1e-3f));
  dl[1] = __logf(fmaxf(fabsf((cyq - cym) / hq), 1e-3f));
  dl[2] = __logf(wq / wm);
  dl[3] = __logf(hq / hm);

  float acch[8] = {0, 0, 0, 0, 0, 0, 0, 0};
#pragma unroll
  for (int j = 0; j < 8; ++j) {
    float invd = __expf(-(float)j * 0.125f * 6.907755278982137f);  // 1000^{-j/8}
#pragma unroll
    for (int d = 0; d < 4; ++d) {
      float ang = 100.0f * dl[d] * invd;
      float s = __sinf(ang), c = __cosf(ang);
      int g = d * 8 + j;
#pragma unroll
      for (int h = 0; h < 8; ++h)
        acch[h] += s * sW[h * 64 + g] + c * sW[h * 64 + 32 + g];
    }
  }
#pragma unroll
  for (int h = 0; h < 8; ++h) {
    float v = acch[h] + sB[h];
    // relu then clamp(1e-6) then log == log(max(v, 1e-6))
    biasOut[((size_t)(b * 8 + h) * 256 + q) * 256 + m] = __logf(fmaxf(v, 1e-6f));
  }
}

// ---------------------------------------------------------------------------
// Attention: one wave per (b, h, 16-query tile).
// scores (16x256) held in registers, softmax via shfl_xor 16-lane reductions,
// probabilities re-fragmented through LDS (bf16) for the P@V WMMAs.
// ---------------------------------------------------------------------------
__global__ __launch_bounds__(32)
void attn_wmma_kernel(const unsigned short* __restrict__ Q,    // [B*H,256,64]
                      const unsigned short* __restrict__ K,    // [B*H,256,64]
                      const unsigned short* __restrict__ Vt,   // [B*H,64,256]
                      const float* __restrict__ biasLog,       // [B*H,256,256]
                      const int* __restrict__ mask,            // [B,256]
                      unsigned short* __restrict__ attnOut) {  // [4096,512]
  __shared__ __align__(16) unsigned short ldsP[16 * 256];
  const int lane = threadIdx.x & 31;
  const int half = lane >> 4;
  const int l16  = lane & 15;
  const int qt = blockIdx.x;  // 0..15 query tile
  const int bh = blockIdx.y;  // 0..127
  const int b = bh >> 3, h = bh & 7;

  const unsigned short* qp = Q + (size_t)bh * 256 * 64;
  const unsigned short* kp = K + (size_t)bh * 256 * 64;
  const unsigned short* vp = Vt + (size_t)bh * 64 * 256;
  const float* bp = biasLog + (size_t)bh * 256 * 256;

  // q A-fragments for the two K-chunks covering d_k = 64
  Frag qf[2];
  {
    const unsigned short* qrow = qp + (size_t)(qt * 16 + l16) * 64;
#pragma unroll
    for (int c = 0; c < 2; ++c) {
#pragma unroll
      for (int v = 0; v < 4; ++v)
        qf[c].u[v] = *(const unsigned int*)(qrow + c * 32 + 8 * half + 2 * v);
#pragma unroll
      for (int v = 0; v < 4; ++v)
        qf[c].u[4 + v] = *(const unsigned int*)(qrow + c * 32 + 16 + 8 * half + 2 * v);
    }
  }

  // --- scores: S = (q * 1/sqrt(dk)) @ k^T --------------------------------
  v8f sc[16];
#pragma unroll 1
  for (int j = 0; j < 16; ++j) {
    Frag kf0, kf1;  // B = k^T: lane column = key j*16 + l16, K dim = d_k
    const unsigned short* krow = kp + (size_t)(j * 16 + l16) * 64;
#pragma unroll
    for (int v = 0; v < 4; ++v) {
      kf0.u[v] = *(const unsigned int*)(krow + 8 * half + 2 * v);
      kf0.u[4 + v] = *(const unsigned int*)(krow + 16 + 8 * half + 2 * v);
      kf1.u[v] = *(const unsigned int*)(krow + 32 + 8 * half + 2 * v);
      kf1.u[4 + v] = *(const unsigned int*)(krow + 48 + 8 * half + 2 * v);
    }
    v8f a{};
    a = wmma_bf16(qf[0], kf0, a);
    a = wmma_bf16(qf[1], kf1, a);
    sc[j] = a;
  }

  // --- mask replace + log-geometric bias ---------------------------------
#pragma unroll 1
  for (int j = 0; j < 16; ++j) {
    int colg = j * 16 + l16;
    int mk = mask[b * 256 + colg];
#pragma unroll
    for (int v = 0; v < 8; ++v) {
      int qrow = qt * 16 + v + 8 * half;
      float s = (mk == 0) ? -1e9f : sc[j][v];
      s += bp[(size_t)qrow * 256 + colg];
      sc[j][v] = s;
    }
  }

  // --- row softmax (row spread over 16 lanes within a half) ---------------
  float inv[8];
#pragma unroll
  for (int v = 0; v < 8; ++v) {
    float mx = sc[0][v];
#pragma unroll
    for (int j = 1; j < 16; ++j) mx = fmaxf(mx, sc[j][v]);
#pragma unroll
    for (int d = 1; d < 16; d <<= 1) mx = fmaxf(mx, __shfl_xor(mx, d, 32));
    float sum = 0.0f;
#pragma unroll
    for (int j = 0; j < 16; ++j) {
      float e = __expf(sc[j][v] - mx);
      sc[j][v] = e;
      sum += e;
    }
#pragma unroll
    for (int d = 1; d < 16; d <<= 1) sum += __shfl_xor(sum, d, 32);
    inv[v] = 1.0f / sum;
  }

  // --- stage P (bf16) in LDS for A-fragment re-layout ---------------------
#pragma unroll 1
  for (int j = 0; j < 16; ++j) {
#pragma unroll
    for (int v = 0; v < 8; ++v) {
      int m = v + 8 * half;
      ldsP[m * 256 + j * 16 + l16] = f2bf(sc[j][v] * inv[v]);
    }
  }
  __syncthreads();

  // --- O = P @ V (4 d-tiles x 8 key chunks) -------------------------------
#pragma unroll 1
  for (int dt = 0; dt < 4; ++dt) {
    v8f o{};
#pragma unroll 1
    for (int c = 0; c < 8; ++c) {
      Frag pf, vf;
      const unsigned short* prow = &ldsP[l16 * 256 + c * 32];
#pragma unroll
      for (int v = 0; v < 4; ++v)
        pf.u[v] = *(const unsigned int*)(prow + 8 * half + 2 * v);
#pragma unroll
      for (int v = 0; v < 4; ++v)
        pf.u[4 + v] = *(const unsigned int*)(prow + 16 + 8 * half + 2 * v);
      const unsigned short* vrow = vp + (size_t)(dt * 16 + l16) * 256 + c * 32;
#pragma unroll
      for (int v = 0; v < 4; ++v) {
        vf.u[v] = *(const unsigned int*)(vrow + 8 * half + 2 * v);
        vf.u[4 + v] = *(const unsigned int*)(vrow + 16 + 8 * half + 2 * v);
      }
      o = wmma_bf16(pf, vf, o);
    }
#pragma unroll
    for (int v = 0; v < 8; ++v) {
      int m = v + 8 * half;
      int r = b * 256 + qt * 16 + m;
      int cc = h * 64 + dt * 16 + l16;
      attnOut[(size_t)r * 512 + cc] = f2bf(o[v]);
    }
  }
}

// ---------------------------------------------------------------------------
extern "C" void kernel_launch(void* const* d_in, const int* in_sizes, int n_in,
                              void* d_out, int out_size, void* d_ws, size_t ws_size,
                              hipStream_t stream) {
  const float* xq  = (const float*)d_in[0];
  const float* xk  = (const float*)d_in[1];
  const float* xv  = (const float*)d_in[2];
  const float* box = (const float*)d_in[3];
  const int*   msk = (const int*)d_in[4];
  const float* Wq  = (const float*)d_in[5];
  const float* bq  = (const float*)d_in[6];
  const float* Wk  = (const float*)d_in[7];
  const float* bk  = (const float*)d_in[8];
  const float* Wv  = (const float*)d_in[9];
  const float* bv  = (const float*)d_in[10];
  const float* Wo  = (const float*)d_in[11];
  const float* bo  = (const float*)d_in[12];
  const float* WGw = (const float*)d_in[13];
  const float* WGb = (const float*)d_in[14];

  char* ws = (char*)d_ws;
  size_t off = 0;
  auto alloc = [&](size_t bytes) {
    char* p = ws + off;
    off += (bytes + 255) & ~(size_t)255;
    return p;
  };
  unsigned short* xqb = (unsigned short*)alloc(4096ull * 512 * 2);
  unsigned short* xkb = (unsigned short*)alloc(4096ull * 512 * 2);
  unsigned short* xvb = (unsigned short*)alloc(4096ull * 512 * 2);
  unsigned short* wqt = (unsigned short*)alloc(512ull * 512 * 2);  // [N,K]
  unsigned short* wkt = (unsigned short*)alloc(512ull * 512 * 2);
  unsigned short* wvt = (unsigned short*)alloc(512ull * 512 * 2);
  unsigned short* wot = (unsigned short*)alloc(512ull * 512 * 2);
  unsigned short* Qb  = (unsigned short*)alloc(4096ull * 512 * 2);  // [B*H,256,64]
  unsigned short* Kb  = (unsigned short*)alloc(4096ull * 512 * 2);  // [B*H,256,64]
  unsigned short* Vtb = (unsigned short*)alloc(4096ull * 512 * 2);  // [B*H,64,256]
  unsigned short* aob = (unsigned short*)alloc(4096ull * 512 * 2);  // attn out bf16
  float* biasF = (float*)alloc(16ull * 8 * 256 * 256 * 4);          // log w_g

  const int CT = 256;
  int nact = 4096 * 512;
  f32_to_bf16_kernel<<<(nact + CT - 1) / CT, CT, 0, stream>>>(xq, xqb, nact);
  f32_to_bf16_kernel<<<(nact + CT - 1) / CT, CT, 0, stream>>>(xk, xkb, nact);
  f32_to_bf16_kernel<<<(nact + CT - 1) / CT, CT, 0, stream>>>(xv, xvb, nact);
  int nw = 512 * 512;
  f32_to_bf16_transpose_kernel<<<(nw + CT - 1) / CT, CT, 0, stream>>>(Wq, wqt);
  f32_to_bf16_transpose_kernel<<<(nw + CT - 1) / CT, CT, 0, stream>>>(Wk, wkt);
  f32_to_bf16_transpose_kernel<<<(nw + CT - 1) / CT, CT, 0, stream>>>(Wv, wvt);
  f32_to_bf16_transpose_kernel<<<(nw + CT - 1) / CT, CT, 0, stream>>>(Wo, wot);

  dim3 gg(32, 256), gb(32);
  gemm_wmma_kernel<<<gg, gb, 0, stream>>>(xqb, wqt, bq, Qb, 0, 0.125f);  // 1/sqrt(64)
  gemm_wmma_kernel<<<gg, gb, 0, stream>>>(xkb, wkt, bk, Kb, 1, 1.0f);
  gemm_wmma_kernel<<<gg, gb, 0, stream>>>(xvb, wvt, bv, Vtb, 2, 1.0f);

  geo_bias_kernel<<<4096, 256, 0, stream>>>(box, WGw, WGb, biasF);

  attn_wmma_kernel<<<dim3(16, 128), 32, 0, stream>>>(Qb, Kb, Vtb, biasF, msk, aob);

  gemm_wmma_kernel<<<gg, gb, 0, stream>>>(aob, wot, bo, d_out, 3, 1.0f);
}
